// DeforAttn_841813590683
// MI455X (gfx1250) — matrix operand order repaired and checked by reading, not compiled, not run on hardware
//
#include <hip/hip_runtime.h>
#include <hip/hip_bf16.h>
#include <math.h>

typedef _Float16 half_t;
typedef __attribute__((ext_vector_type(16))) _Float16 v16h;
typedef __attribute__((ext_vector_type(8)))  _Float16 v8h;
typedef __attribute__((ext_vector_type(8)))  float    v8f;

#define BS_   8
#define NQ_   4096
#define C_    256
#define NH_   8
#define NP_   4
#define L_    6
#define NV_   10752
#define MQ_   (BS_*NQ_)    /* 32768 */
#define MV_   (BS_*NV_)    /* 86016 */
#define NOFF_ 384
#define NATT_ 192

// ---------------------------------------------------------------- prep ------
__global__ void k_add_cvt(const float* __restrict__ a, const float* __restrict__ b,
                          half_t* __restrict__ o, int n) {
  int i = blockIdx.x * blockDim.x + threadIdx.x;
  if (i < n) o[i] = (half_t)(a[i] + b[i]);
}

__global__ void k_cvt(const float* __restrict__ a, half_t* __restrict__ o, int n) {
  int i = blockIdx.x * blockDim.x + threadIdx.x;
  if (i < n) o[i] = (half_t)a[i];
}

// Pack row-major f32 W[K,N] into WMMA 16x16x32 B-fragment order:
// flat i = (((kt*tilesN + nt)*32 + lane)*16 + idx)
//   element = W[kt*32 + (idx>>3)*16 + (lane>>4)*8 + (idx&7)][nt*16 + (lane&15)]
__global__ void k_pack_w(const float* __restrict__ W, half_t* __restrict__ Bp,
                         int N, int total) {
  int i = blockIdx.x * blockDim.x + threadIdx.x;
  if (i >= total) return;
  int idx    = i & 15;
  int lane   = (i >> 4) & 31;
  int tile   = i >> 9;
  int tilesN = N >> 4;
  int kt = tile / tilesN;
  int nt = tile - kt * tilesN;
  int k = kt * 32 + (idx >> 3) * 16 + (lane >> 4) * 8 + (idx & 7);
  int n = nt * 16 + (lane & 15);
  Bp[i] = (half_t)W[(size_t)k * N + n];
}

// ------------------------------------------------------------ WMMA GEMM ----
// C[M,N] = A[M,K](f16, row-major) x Bp(pre-packed f16) + bias, (+ residual)
// One wave computes a 16 x (16*NTG) strip: the A fragment is loaded once per
// k-step and reused across NTG consecutive N-tiles (NTG WMMAs per A-load),
// cutting A-operand HBM traffic by NTG vs one-tile-per-wave.
// K multiple of 32, M multiple of 16, tilesN multiple of NTG.
template <int NTG>
__global__ void k_gemm_wmma(const half_t* __restrict__ A, const half_t* __restrict__ Bp,
                            const float* __restrict__ bias, const float* __restrict__ residual,
                            float* __restrict__ Cf, half_t* __restrict__ Ch,
                            int M, int N, int K) {
  int gid  = blockIdx.x * blockDim.x + threadIdx.x;
  int wave = gid >> 5;
  int lane = threadIdx.x & 31;
  int tilesN = N >> 4;
  int groups = tilesN / NTG;
  int totalWork = (M >> 4) * groups;
  if (wave >= totalWork) return;
  int tm  = wave / groups;
  int g   = wave - tm * groups;
  int tn0 = g * NTG;
  int hi16 = lane >> 4;        // 0 or 1 (K-half select for A, C row-half)
  int l15  = lane & 15;        // A row / B,C column

  // A fragment: lane(0-15)=row M, halfs [0..7] = K (hi16*8 .. +7),
  //             halfs [8..15] = K (16 + hi16*8 .. +7)
  const half_t* Ap = A + (size_t)(tm * 16 + l15) * K + hi16 * 8;
  v8f acc[NTG] = {};
  for (int kt = 0; kt < (K >> 5); ++kt) {
    v8h alo = *(const v8h*)(Ap + kt * 32);
    v8h ahi = *(const v8h*)(Ap + kt * 32 + 16);
    v16h a = __builtin_shufflevector(alo, ahi, 0,1,2,3,4,5,6,7,8,9,10,11,12,13,14,15);
    // prefetch the next k-slice of the streamed A operand (near scope)
    __builtin_prefetch(Ap + kt * 32 + 32, 0, 3);
    #pragma unroll
    for (int j = 0; j < NTG; ++j) {
      const half_t* bp = Bp + (((size_t)kt * tilesN + tn0 + j) * 32 + lane) * 16;
      v8h blo = *(const v8h*)(bp);
      v8h bhi = *(const v8h*)(bp + 8);
      v16h b = __builtin_shufflevector(blo, bhi, 0,1,2,3,4,5,6,7,8,9,10,11,12,13,14,15);
      acc[j] = __builtin_amdgcn_wmma_f32_16x16x32_f16(
          /*neg_a=*/false, a, /*neg_b=*/false, b,
          /*c_mod=*/(short)0, acc[j], /*reuse_a=*/false, /*reuse_b=*/false);
    }
  }

  #pragma unroll
  for (int j = 0; j < NTG; ++j) {
    int n = (tn0 + j) * 16 + l15;
    float bv = bias ? bias[n] : 0.0f;
    #pragma unroll
    for (int r = 0; r < 8; ++r) {         // C/D: VGPR r -> row (hi16*8 + r)
      int m = tm * 16 + hi16 * 8 + r;
      size_t idx = (size_t)m * N + n;
      float v = acc[j][r] + bv;
      if (Ch) Ch[idx] = (half_t)v;
      else    Cf[idx] = residual ? (v + residual[idx]) : v;
    }
  }
}

// ------------------------------------------------------------- softmax -----
// attn layout: [bq][NH*24], softmax over each contiguous group of 24.
__global__ void k_softmax24(float* __restrict__ attn, int total) {
  int t = blockIdx.x * blockDim.x + threadIdx.x;
  if (t >= total) return;
  float* p = attn + (size_t)t * (L_ * NP_);
  float m = -3.4e38f;
  #pragma unroll
  for (int i = 0; i < L_ * NP_; ++i) m = fmaxf(m, p[i]);
  float s = 0.f;
  float e[L_ * NP_];
  #pragma unroll
  for (int i = 0; i < L_ * NP_; ++i) { e[i] = expf(p[i] - m); s += e[i]; }
  float inv = 1.0f / s;
  #pragma unroll
  for (int i = 0; i < L_ * NP_; ++i) p[i] = e[i] * inv;
}

// ------------------------------------------------------------- sampling ----
__device__ __forceinline__ float tap(const half_t* __restrict__ vb,
                                     int s0, int H, int W, int y, int x) {
  if (y < 0 || y >= H || x < 0 || x >= W) return 0.0f;   // zero padding
  return (float)vb[(size_t)(s0 + y * W + x) * C_];
}

// One wave per (b, q, h); lane = channel 0..31 (d = C/NH = 32).
__global__ void k_sample(const half_t* __restrict__ vproj, const float* __restrict__ ref,
                         const float* __restrict__ off, const float* __restrict__ attn,
                         half_t* __restrict__ sampled) {
  const int Hs[L_] = {64, 32, 16, 64, 32, 16};
  const int Ws[L_] = {64, 32, 16, 64, 32, 16};
  const int S0[L_] = {0, 4096, 5120, 5376, 9472, 10496};
  int gid  = blockIdx.x * blockDim.x + threadIdx.x;
  int wid  = gid >> 5;
  int lane = threadIdx.x & 31;
  int h  = wid & (NH_ - 1);
  int bq = wid >> 3;               // b*NQ + q
  int b  = bq >> 12;               // NQ = 4096
  const float* refp = ref  + (size_t)bq * (L_ * 2);
  const float* offp = off  + (size_t)bq * NOFF_ + h * (L_ * NP_ * 2);
  const float* atp  = attn + (size_t)bq * NATT_ + h * (L_ * NP_);
  const half_t* vb  = vproj + (size_t)b * NV_ * C_ + h * 32 + lane;
  float acc = 0.0f;
  #pragma unroll
  for (int l = 0; l < L_; ++l) {
    const int H = Hs[l], W = Ws[l], s0 = S0[l];
    float rx = refp[l * 2 + 0] * (float)W - 0.5f;   // ref*W + off - 0.5
    float ry = refp[l * 2 + 1] * (float)H - 0.5f;
    #pragma unroll
    for (int p = 0; p < NP_; ++p) {
      float x = rx + offp[(l * NP_ + p) * 2 + 0];
      float y = ry + offp[(l * NP_ + p) * 2 + 1];
      float w = atp[l * NP_ + p];
      float xf = floorf(x), yf = floorf(y);
      int x0 = (int)xf, y0 = (int)yf;
      float lx = x - xf, ly = y - yf;
      float w00 = (1.f - lx) * (1.f - ly) * w;
      float w01 = lx * (1.f - ly) * w;
      float w10 = (1.f - lx) * ly * w;
      float w11 = lx * ly * w;
      acc += w00 * tap(vb, s0, H, W, y0,     x0);
      acc += w01 * tap(vb, s0, H, W, y0,     x0 + 1);
      acc += w10 * tap(vb, s0, H, W, y0 + 1, x0);
      acc += w11 * tap(vb, s0, H, W, y0 + 1, x0 + 1);
    }
  }
  sampled[(size_t)bq * C_ + h * 32 + lane] = (half_t)acc;
}

// ------------------------------------------------------------- launch ------
extern "C" void kernel_launch(void* const* d_in, const int* in_sizes, int n_in,
                              void* d_out, int out_size, void* d_ws, size_t ws_size,
                              hipStream_t stream) {
  const float* query     = (const float*)d_in[0];
  const float* query_pos = (const float*)d_in[1];
  const float* value     = (const float*)d_in[2];
  const float* refpts    = (const float*)d_in[3];
  // d_in[4] = spatial_shapes (static, hardcoded)
  const float* W_off  = (const float*)d_in[5];
  const float* b_off  = (const float*)d_in[6];
  const float* W_attn = (const float*)d_in[7];
  const float* b_attn = (const float*)d_in[8];
  const float* W_val  = (const float*)d_in[9];
  const float* b_val  = (const float*)d_in[10];
  const float* W_out  = (const float*)d_in[11];
  const float* b_out  = (const float*)d_in[12];
  float* out = (float*)d_out;

  // workspace carve-out (256B aligned)
  char* ws = (char*)d_ws;
  size_t o = 0;
  auto carve = [&](size_t bytes) { char* p = ws + o; o += (bytes + 255) & ~(size_t)255; return p; };
  half_t* q16      = (half_t*)carve((size_t)MQ_ * C_ * 2);      // 16.8 MB
  half_t* v16      = (half_t*)carve((size_t)MV_ * C_ * 2);      // 44.0 MB
  half_t* vproj16  = (half_t*)carve((size_t)MV_ * C_ * 2);      // 44.0 MB
  float*  offbuf   = (float*)carve((size_t)MQ_ * NOFF_ * 4);    // 50.3 MB
  float*  attnbuf  = (float*)carve((size_t)MQ_ * NATT_ * 4);    // 25.2 MB
  half_t* sampled  = (half_t*)carve((size_t)MQ_ * C_ * 2);      // 16.8 MB
  half_t* Wval_p   = (half_t*)carve((size_t)C_ * C_ * 2);
  half_t* Woff_p   = (half_t*)carve((size_t)C_ * NOFF_ * 2);
  half_t* Wattn_p  = (half_t*)carve((size_t)C_ * NATT_ * 2);
  half_t* Wout_p   = (half_t*)carve((size_t)C_ * C_ * 2);

  const int TPB = 256;
  // 1) q = half(query + query_pos); v16 = half(value)
  k_add_cvt<<<(MQ_ * C_) / TPB, TPB, 0, stream>>>(query, query_pos, q16, MQ_ * C_);
  k_cvt<<<(MV_ * C_) / TPB, TPB, 0, stream>>>(value, v16, MV_ * C_);
  // 2) pack weights into B-fragment layout
  k_pack_w<<<(C_ * C_) / TPB, TPB, 0, stream>>>(W_val, Wval_p, C_, C_ * C_);
  k_pack_w<<<(C_ * NOFF_) / TPB, TPB, 0, stream>>>(W_off, Woff_p, NOFF_, C_ * NOFF_);
  k_pack_w<<<(C_ * NATT_) / TPB, TPB, 0, stream>>>(W_attn, Wattn_p, NATT_, C_ * NATT_);
  k_pack_w<<<(C_ * C_) / TPB, TPB, 0, stream>>>(W_out, Wout_p, C_, C_ * C_);
  // 3) value projection GEMM -> f16 [MV, 256]  (tilesN=16, NTG=8 -> 2 groups)
  {
    int waves = (MV_ / 16) * (C_ / 16 / 8);
    k_gemm_wmma<8><<<(waves * 32 + TPB - 1) / TPB, TPB, 0, stream>>>(
        v16, Wval_p, b_val, nullptr, nullptr, vproj16, MV_, C_, C_);
  }
  // 4) offsets GEMM -> f32 [MQ, 384]  (tilesN=24, NTG=8 -> 3 groups)
  {
    int waves = (MQ_ / 16) * (NOFF_ / 16 / 8);
    k_gemm_wmma<8><<<(waves * 32 + TPB - 1) / TPB, TPB, 0, stream>>>(
        q16, Woff_p, b_off, nullptr, offbuf, nullptr, MQ_, NOFF_, C_);
  }
  // 5) attention GEMM -> f32 [MQ, 192]  (tilesN=12, NTG=6 -> 2 groups)
  {
    int waves = (MQ_ / 16) * (NATT_ / 16 / 6);
    k_gemm_wmma<6><<<(waves * 32 + TPB - 1) / TPB, TPB, 0, stream>>>(
        q16, Wattn_p, b_attn, nullptr, attnbuf, nullptr, MQ_, NATT_, C_);
  }
  k_softmax24<<<(MQ_ * NH_) / TPB, TPB, 0, stream>>>(attnbuf, MQ_ * NH_);
  // 6) deformable sampling -> f16 [MQ, 256]
  k_sample<<<(MQ_ * NH_ * 32) / TPB, TPB, 0, stream>>>(
      vproj16, refpts, offbuf, attnbuf, sampled);
  // 7) output GEMM + bias + residual(query) -> f32 d_out  (NTG=8 -> 2 groups)
  {
    int waves = (MQ_ / 16) * (C_ / 16 / 8);
    k_gemm_wmma<8><<<(waves * 32 + TPB - 1) / TPB, TPB, 0, stream>>>(
        sampled, Wout_p, b_out, query, out, nullptr, MQ_, C_, C_);
  }
  (void)in_sizes; (void)n_in; (void)out_size; (void)ws_size;
}
